// PrototypeHeadKMeans_73985106641559
// MI455X (gfx1250) — matrix operand order repaired and checked by reading, not compiled, not run on hardware
//
#include <hip/hip_runtime.h>
#include <hip/hip_bf16.h>
#include <math.h>

typedef __attribute__((ext_vector_type(16))) __bf16 v16bf;
typedef __attribute__((ext_vector_type(8)))  float  v8f;

union FragBF { v16bf v; unsigned int u[8]; };

#define EDIM 256
#define KCLUS 8
#define NCLS 6
#define NPAIR 3            // class pairs: 2 classes of 8 clusters per 16-col B tile
#define KITERS 10

// ---------------------------------------------------------------- mu ----
__global__ __launch_bounds__(256) void mu_partial_kernel(
    const float* __restrict__ feats, const unsigned char* __restrict__ valid,
    float* __restrict__ muAcc, float* __restrict__ cntAcc) {
  int d = threadIdx.x;                       // dim
  size_t base = (size_t)blockIdx.x * 256;    // 256 rows per block
  float s = 0.0f, cnt = 0.0f;
  for (int r = 0; r < 256; ++r) {
    size_t gr = base + r;
    float m = valid[gr] ? 1.0f : 0.0f;
    s += feats[gr * EDIM + d] * m;
    if (d == 0) cnt += m;
  }
  atomicAdd(&muAcc[d], s);
  if (d == 0) atomicAdd(cntAcc, cnt);
}

__global__ __launch_bounds__(256) void mu_final_kernel(
    const float* __restrict__ muAcc, const float* __restrict__ cntAcc,
    float* __restrict__ mu) {
  int d = threadIdx.x;
  mu[d] = muAcc[d] / fmaxf(*cntAcc, 1e-6f);
}

// --------------------------------------------------------- inv norms ----
__global__ __launch_bounds__(256) void invnorm_kernel(
    const float* __restrict__ src, const float* __restrict__ mu,
    const unsigned char* __restrict__ valid, float* __restrict__ invOut) {
  int row  = blockIdx.x * 8 + (threadIdx.x >> 5);   // one wave32 per row
  int lane = threadIdx.x & 31;
  float ss = 0.0f;
  size_t rb = (size_t)row * EDIM;
  for (int d = lane; d < EDIM; d += 32) {
    float x = src[rb + d] - mu[d];
    ss += x * x;
  }
  for (int o = 16; o > 0; o >>= 1) ss += __shfl_xor(ss, o, 32);
  if (lane == 0) {
    float inv = 1.0f / fmaxf(sqrtf(ss), 1e-12f);
    if (valid && valid[row] == 0) inv = 0.0f;       // feats *= m after norm
    invOut[row] = inv;
  }
}

// ------------------------------------------------ centroid init ---------
__global__ __launch_bounds__(32) void init_cent_kernel(
    const float* __restrict__ feats, const float* __restrict__ labels,
    const unsigned char* __restrict__ valid, const float* __restrict__ mu,
    const float* __restrict__ inv_s, float* __restrict__ cent, int nRows) {
  int c = blockIdx.x, lane = threadIdx.x;
  __shared__ int idxS[KCLUS];
  __shared__ int foundS;
  if (lane == 0) foundS = 0;
  __syncthreads();
  for (int base = 0; base < nRows; base += 32) {
    if (foundS >= KCLUS) break;                      // uniform (post-barrier)
    int gr = base + lane;
    int pred = (labels[(size_t)gr * NCLS + c] > 0.5f) && (valid[gr] != 0);
    unsigned int bal = (unsigned int)__ballot(pred); // wave32: low 32 bits
    if (lane == 0) {
      int f = foundS;
      while (bal && f < KCLUS) {
        idxS[f++] = base + __builtin_ctz(bal);
        bal &= bal - 1;
      }
      foundS = f;
    }
    __syncthreads();
  }
  __syncthreads();
  int found = foundS;
  for (int i = lane; i < KCLUS * EDIM; i += 32) {
    int k = i >> 8, d = i & 255;
    float v = 0.0f;
    if (k < found) {
      int r = idxS[k];
      v = (feats[(size_t)r * EDIM + d] - mu[d]) * inv_s[r];
    }
    cent[(size_t)(c * KCLUS + k) * EDIM + d] = v;
  }
}

// ------------------- k-means assign + gather (2 classes per block) ------
__global__ __launch_bounds__(128) void assign_kernel(
    const float* __restrict__ feats, const float* __restrict__ labels,
    const unsigned char* __restrict__ valid, const float* __restrict__ mu,
    const float* __restrict__ inv_s, const float* __restrict__ cent,
    float* __restrict__ sums, float* __restrict__ counts) {
  __shared__ __bf16 As[64][EDIM];        // 32 KB: 64 normalized points
  __shared__ __bf16 Bs[16][EDIM];        //  8 KB: 16 centroids (classes c0,c1)
  __shared__ float  muS[EDIM];           //  1 KB
  __shared__ float  simsS[64][16];       //  4 KB
  __shared__ float  sumsS[16 * EDIM];    // 16 KB: per-pair partial sums
  __shared__ float  cntS[16];
  __shared__ int    asgS[2][64];
  __shared__ float  wS[2][64];

  int pair = blockIdx.y;                 // classes 2*pair, 2*pair+1
  int rowBase = blockIdx.x * 64;
  int tid = threadIdx.x;

  for (int d = tid; d < EDIM; d += 128) muS[d] = mu[d];
  for (int i = tid; i < 16 * EDIM; i += 128) sumsS[i] = 0.0f;
  if (tid < 16) cntS[tid] = 0.0f;
  __syncthreads();

  for (int i = tid; i < 64 * EDIM; i += 128) {
    int r = i >> 8, d = i & 255;
    int gr = rowBase + r;
    float x = (feats[(size_t)gr * EDIM + d] - muS[d]) * inv_s[gr];
    As[r][d] = (__bf16)x;
  }
  // 16 contiguous centroid rows of this class pair ([C][K][E] flat)
  for (int i = tid; i < 16 * EDIM; i += 128) {
    int r = i >> 8, d = i & 255;
    Bs[r][d] = (__bf16)cent[(size_t)(pair * 16 + r) * EDIM + d];
  }
  __syncthreads();

  // Each wave: 16(point) x 16(cluster-slot) tile, K=256 in 8 chunks of 32.
  int wv = tid >> 5, lane = tid & 31;
  int ml = lane & 15, hi = lane >> 4;
  v8f acc = {0.f, 0.f, 0.f, 0.f, 0.f, 0.f, 0.f, 0.f};
  const unsigned int* Arow = reinterpret_cast<const unsigned int*>(&As[wv * 16 + ml][0]);
  const unsigned int* Brow = reinterpret_cast<const unsigned int*>(&Bs[ml][0]);
  for (int kc = 0; kc < 8; ++kc) {
    FragBF a, b;
#pragma unroll
    for (int v = 0; v < 8; ++v) {
      int ka = kc * 32 + (hi << 4) + (v << 1);                        // contiguous k layout (matches B)
      a.u[v] = Arow[ka >> 1];
      int kb = kc * 32 + (hi << 4) + (v << 1);                        // contiguous k layout (matches A)
      b.u[v] = Brow[kb >> 1];
    }
    acc = __builtin_amdgcn_wmma_f32_16x16x32_bf16(
        false, a.v, false, b.v, (short)0, acc, false, false);
  }
#pragma unroll
  for (int j = 0; j < 8; ++j) simsS[wv * 16 + hi * 8 + j][ml] = acc[j];
  __syncthreads();

  if (tid < 64) {
    int p = tid;
    int gr = rowBase + p;
#pragma unroll
    for (int h = 0; h < 2; ++h) {                       // two classes per block
      float best = simsS[p][h * KCLUS];
      int bi = 0;
#pragma unroll
      for (int j = 1; j < KCLUS; ++j) {
        float s = simsS[p][h * KCLUS + j];
        if (s > best) { best = s; bi = j; }             // first-max tiebreak
      }
      asgS[h][p] = bi;
      int c = pair * 2 + h;
      float w = ((labels[(size_t)gr * NCLS + c] > 0.5f) && (valid[gr] != 0)) ? 1.0f : 0.0f;
      wS[h][p] = w;
      if (w != 0.0f) atomicAdd(&cntS[h * KCLUS + bi], 1.0f);
    }
  }
  __syncthreads();

  for (int p = 0; p < 64; ++p) {
#pragma unroll
    for (int h = 0; h < 2; ++h) {
      if (wS[h][p] != 0.0f) {
        int a = h * KCLUS + asgS[h][p];
        for (int d = tid; d < EDIM; d += 128)
          atomicAdd(&sumsS[a * EDIM + d], (float)As[p][d]);
      }
    }
  }
  __syncthreads();

  for (int i = tid; i < 16 * EDIM; i += 128)
    atomicAdd(&sums[(size_t)pair * 16 * EDIM + i], sumsS[i]);
  if (tid < 16) {
    float cc = cntS[tid];
    if (cc != 0.0f) atomicAdd(&counts[pair * 16 + tid], cc);
  }
}

// ------------------------------------------------ centroid update -------
__global__ __launch_bounds__(256) void update_kernel(
    float* __restrict__ cent, float* __restrict__ sums, float* __restrict__ counts) {
  int c = blockIdx.x, tid = threadIdx.x;
  __shared__ float red[256];
  for (int k = 0; k < KCLUS; ++k) {
    float cnt = counts[c * KCLUS + k];
    size_t idx = (size_t)(c * KCLUS + k) * EDIM + tid;
    float v = (cnt > 0.0f) ? (sums[idx] / fmaxf(cnt, 1.0f)) : cent[idx];
    sums[idx] = 0.0f;                                    // re-zero for next iter
    red[tid] = v * v;
    __syncthreads();
    for (int s = 128; s > 0; s >>= 1) {
      if (tid < s) red[tid] += red[tid + s];
      __syncthreads();
    }
    float nrm = red[0];
    __syncthreads();
    cent[idx] = v / fmaxf(sqrtf(nrm), 1e-12f);
  }
  if (tid < KCLUS) counts[c * KCLUS + tid] = 0.0f;
}

// ---------------------------------------------------- query scoring -----
__global__ __launch_bounds__(64) void score_kernel(
    const float* __restrict__ qf, const float* __restrict__ mu,
    const float* __restrict__ inv_q, const float* __restrict__ cent,
    float* __restrict__ out) {
  __shared__ __bf16 Aq[32][EDIM];            // 16 KB: 32 queries
  __shared__ __bf16 Bs[NCLS * KCLUS][EDIM];  // 24 KB: all 48 centroids
  __shared__ float  muS[EDIM];
  __shared__ float  simsS[32][NCLS * KCLUS]; //  6 KB

  int rowBase = blockIdx.x * 32;
  int tid = threadIdx.x;                     // 64 threads = 2 waves

  for (int d = tid; d < EDIM; d += 64) muS[d] = mu[d];
  __syncthreads();
  for (int i = tid; i < 32 * EDIM; i += 64) {
    int r = i >> 8, d = i & 255;
    int gq = rowBase + r;
    Aq[r][d] = (__bf16)((qf[(size_t)gq * EDIM + d] - muS[d]) * inv_q[gq]);
  }
  for (int i = tid; i < NCLS * KCLUS * EDIM; i += 64) {
    int r = i >> 8, d = i & 255;
    Bs[r][d] = (__bf16)cent[(size_t)r * EDIM + d];
  }
  __syncthreads();

  int wv = tid >> 5, lane = tid & 31;
  int ml = lane & 15, hi = lane >> 4;
  const unsigned int* Arow = reinterpret_cast<const unsigned int*>(&Aq[wv * 16 + ml][0]);
  for (int t = 0; t < 3; ++t) {              // 48 centroids = 3 N-tiles of 16
    const unsigned int* Brow = reinterpret_cast<const unsigned int*>(&Bs[t * 16 + ml][0]);
    v8f acc = {0.f, 0.f, 0.f, 0.f, 0.f, 0.f, 0.f, 0.f};
    for (int kc = 0; kc < 8; ++kc) {
      FragBF a, b;
#pragma unroll
      for (int v = 0; v < 8; ++v) {
        int ka = kc * 32 + (hi << 4) + (v << 1);
        a.u[v] = Arow[ka >> 1];
        int kb = kc * 32 + (hi << 4) + (v << 1);
        b.u[v] = Brow[kb >> 1];
      }
      acc = __builtin_amdgcn_wmma_f32_16x16x32_bf16(
          false, a.v, false, b.v, (short)0, acc, false, false);
    }
#pragma unroll
    for (int j = 0; j < 8; ++j) simsS[wv * 16 + hi * 8 + j][t * 16 + ml] = acc[j];
  }
  __syncthreads();

  if (tid < 32) {
    int p = tid, gq = rowBase + p;
    for (int c = 0; c < NCLS; ++c) {
      float m = -1e30f;
#pragma unroll
      for (int j = 0; j < KCLUS; ++j) m = fmaxf(m, simsS[p][c * KCLUS + j]);
      float s = 0.0f;
#pragma unroll
      for (int j = 0; j < KCLUS; ++j)
        s += expf(20.0f * simsS[p][c * KCLUS + j] - 20.0f * m);
      // scores * temp == logsumexp(ALPHA*sims) since ALPHA == temp == 20
      out[(size_t)gq * NCLS + c] = 20.0f * m + logf(s);
    }
  }
}

// ---------------------------------------------------------------- host --
extern "C" void kernel_launch(void* const* d_in, const int* in_sizes, int n_in,
                              void* d_out, int out_size, void* d_ws, size_t ws_size,
                              hipStream_t stream) {
  const float*         feats  = (const float*)d_in[0];
  const float*         labels = (const float*)d_in[1];
  const unsigned char* valid  = (const unsigned char*)d_in[2];
  const float*         qf     = (const float*)d_in[3];
  float*               out    = (float*)d_out;

  int N  = in_sizes[0] / EDIM;   // 131072 support rows
  int NQ = in_sizes[3] / EDIM;   // 131072 query rows

  char* ws = (char*)d_ws;
  float* muAcc  = (float*)(ws + 0);                              // 256 f
  float* cntAcc = (float*)(ws + 1024);                           // 1 f
  float* mu     = (float*)(ws + 2048);                           // 256 f
  float* inv_s  = (float*)(ws + 4096);                           // N f
  float* inv_q  = (float*)(ws + 4096 + (size_t)N * 4);           // NQ f
  float* cent   = (float*)(ws + 4096 + (size_t)(N + NQ) * 4);    // 48*256 f
  float* sums   = cent + NCLS * KCLUS * EDIM;                    // 48*256 f
  float* counts = sums + NCLS * KCLUS * EDIM;                    // 48 f
  size_t wsUsed = 4096 + (size_t)(N + NQ) * 4 +
                  (size_t)(2 * NCLS * KCLUS * EDIM + NCLS * KCLUS) * 4;

  hipMemsetAsync(d_ws, 0, wsUsed, stream);

  mu_partial_kernel<<<N / 256, 256, 0, stream>>>(feats, valid, muAcc, cntAcc);
  mu_final_kernel<<<1, 256, 0, stream>>>(muAcc, cntAcc, mu);

  invnorm_kernel<<<N / 8, 256, 0, stream>>>(feats, mu, valid, inv_s);
  invnorm_kernel<<<NQ / 8, 256, 0, stream>>>(qf, mu, nullptr, inv_q);

  init_cent_kernel<<<NCLS, 32, 0, stream>>>(feats, labels, valid, mu, inv_s, cent, N);

  for (int it = 0; it < KITERS; ++it) {
    assign_kernel<<<dim3(N / 64, NPAIR), 128, 0, stream>>>(
        feats, labels, valid, mu, inv_s, cent, sums, counts);
    update_kernel<<<NCLS, 256, 0, stream>>>(cent, sums, counts);
  }

  score_kernel<<<NQ / 32, 64, 0, stream>>>(qf, mu, inv_q, cent, out);
}
